// GVPDynamicProjection_38689065402610
// MI455X (gfx1250) — matrix operand order repaired and checked by reading, not compiled, not run on hardware
//
#include <hip/hip_runtime.h>
#include <math.h>

typedef __attribute__((ext_vector_type(16))) _Float16 v16h;
typedef __attribute__((ext_vector_type(8)))  float    v8f;
typedef _Float16 half_t;

#define INL __device__ __forceinline__

#define N_NODES 8192
#define CH      16
#define NCROWS  (N_NODES*CH)   /* 131072 */

INL v8f wmma_f16(v16h a, v16h b, v8f c) {
  return __builtin_amdgcn_wmma_f32_16x16x32_f16(false, a, false, b, (short)0, c, false, false);
}

INL void lds_fence() { asm volatile("s_wait_dscnt 0" ::: "memory"); }
INL int lane_id() { return (int)(threadIdx.x & 31u); }

INL unsigned fenc(float f) { unsigned u = __float_as_uint(f); return (u & 0x80000000u) ? ~u : (u | 0x80000000u); }
INL float    fdec(unsigned e) { unsigned u = (e & 0x80000000u) ? (e ^ 0x80000000u) : ~e; return __uint_as_float(u); }

// ---------------- fragment loaders (CDNA5 wave32 16x16x32 f16 layouts) ----------------
// A[m][k] from row-major f16, row stride ld (no converts)
INL v16h loadA_f16(const half_t* p, int ld, int m0, int k0) {
  int l = lane_id();
  int m = m0 + (l & 15);
  int kb = k0 + ((l & 16) ? 16 : 0);
  v16h a;
#pragma unroll
  for (int j = 0; j < 8; ++j) {
    a[j]     = p[m*ld + kb + j];
    a[j + 8] = p[m*ld + kb + 8 + j];
  }
  return a;
}

// A[m][k] from f32 storage (convert on load)
INL v16h loadA_f32(const float* p, int ld, int m0, int k0) {
  int l = lane_id();
  int m = m0 + (l & 15);
  int kb = k0 + ((l & 16) ? 16 : 0);
  v16h a;
#pragma unroll
  for (int j = 0; j < 8; ++j) {
    a[j]     = (_Float16)p[m*ld + kb + j];
    a[j + 8] = (_Float16)p[m*ld + kb + 8 + j];
  }
  return a;
}

// A[m][k] = p[k*ld + m] (transposed f16 storage)
INL v16h loadA_f16T(const half_t* p, int ld, int m0, int k0) {
  int l = lane_id();
  int m = m0 + (l & 15);
  int kb = k0 + ((l & 16) ? 16 : 0);
  v16h a;
#pragma unroll
  for (int j = 0; j < 8; ++j) {
    a[j]     = p[(kb + j)*ld + m];
    a[j + 8] = p[(kb + 8 + j)*ld + m];
  }
  return a;
}

// B[k][n] = W[n][k]; W row-major [nout][kin] f16 (computes X @ W^T)
INL v16h loadB_Wt(const half_t* w, int ld, int n0, int k0) {
  int l = lane_id();
  int n = n0 + (l & 15);
  int kb = k0 + ((l & 16) ? 16 : 0);
  v16h b;
#pragma unroll
  for (int j = 0; j < 16; ++j) b[j] = w[n*ld + kb + j];
  return b;
}

// B[k][n] from row-major [k][n] f16 storage
INL v16h loadB_kn(const half_t* p, int ld, int k0, int n0) {
  int l = lane_id();
  int n = n0 + (l & 15);
  int kb = k0 + ((l & 16) ? 16 : 0);
  v16h b;
#pragma unroll
  for (int j = 0; j < 16; ++j) b[j] = p[(kb + j)*ld + n];
  return b;
}

INL void storeD(float* p, int ld, int m0, int n0, v8f d) {
  int l = lane_id();
  int n = n0 + (l & 15);
  int mb = m0 + ((l & 16) ? 8 : 0);
#pragma unroll
  for (int i = 0; i < 8; ++i) p[(mb + i)*ld + n] = d[i];
}

INL void storeD_bias(float* p, int ld, int m0, int n0, v8f d, const float* bias) {
  int l = lane_id();
  int n = n0 + (l & 15);
  int mb = m0 + ((l & 16) ? 8 : 0);
  float bn = bias[n];
#pragma unroll
  for (int i = 0; i < 8; ++i) p[(mb + i)*ld + n] = d[i] + bn;
}

// ---------------- GVP helpers (one wave = 16 rows) ----------------
// stage s rows -> sXh[16][128] f16; v rows -> vth[48][32] f16 (row = r*3+d, col = vi)
INL void stage_sv(const float* s_in, const float* v_in, const float* Zbuf,
                  int row0, half_t* sXh, half_t* vth) {
  for (int i = lane_id(); i < 16*128; i += 32) {
    int r = i >> 7;
    float sc = 1.0f;
    if (Zbuf) { int row = row0 + r; sc = 1.0f / Zbuf[(row & 15)*32 + (row >> 4)]; }
    sXh[i] = (_Float16)(s_in[(size_t)(row0 + r)*128 + (i & 127)] * sc);
  }
  for (int i = lane_id(); i < 16*96; i += 32) {
    int r = i / 96, f = i - r*96;
    int vi = f / 3, d = f - vi*3;
    float sc = 1.0f;
    if (Zbuf) { int row = row0 + r; sc = 1.0f / Zbuf[(row & 15)*32 + (row >> 4)]; }
    vth[(r*3 + d)*32 + vi] = (_Float16)(v_in[(size_t)(row0 + r)*96 + f] * sc);
  }
  lds_fence();
}

// vh[48][32] = vt @ wh^T (f32) ; vnh[16][32] = sqrt(max(sum_d vh^2, eps)) (f16);
// optional vh16 = f16 copy of vh (A operand for the v_out GEMM)
INL void compute_vh_vn(const half_t* vth, const half_t* wh, float* vh,
                       half_t* vnh, half_t* vh16) {
#pragma unroll
  for (int mt = 0; mt < 3; ++mt) {
    v16h a = loadA_f16(vth, 32, mt*16, 0);
#pragma unroll
    for (int nt = 0; nt < 2; ++nt) {
      v8f acc = {};
      acc = wmma_f16(a, loadB_Wt(wh, 32, nt*16, 0), acc);
      storeD(vh, 32, mt*16, nt*16, acc);
    }
  }
  lds_fence();
  for (int i = lane_id(); i < 16*32; i += 32) {
    int r = i >> 5, h = i & 31;
    float x0 = vh[(r*3 + 0)*32 + h];
    float x1 = vh[(r*3 + 1)*32 + h];
    float x2 = vh[(r*3 + 2)*32 + h];
    vnh[i] = (_Float16)sqrtf(fmaxf(x0*x0 + x1*x1 + x2*x2, 1e-8f));
  }
  if (vh16)
    for (int i = lane_id(); i < 48*32; i += 32) vh16[i] = (_Float16)vh[i];
  lds_fence();
}

// ---------------- kernels ----------------
__global__ void k_cvt(const float* __restrict__ src, half_t* __restrict__ dst, int n) {
  int i = blockIdx.x*256 + threadIdx.x;
  if (i < n) dst[i] = (_Float16)src[i];
}

__global__ void k_initmax(unsigned* __restrict__ gmax) {
  int i = blockIdx.x*256 + threadIdx.x;
  if (i < 512) gmax[i] = 0x007FFFFFu;   // enc(-inf)
}

// per-wave region: vh f32 [48][32], then f16: sXh[16][128], vth[48][32], vnh[16][32]
#define SCALAR_WAVE_BYTES (48*32*4 + (16*128 + 48*32 + 16*32)*2)   /* 14336 */

// weight_proj GVP -> logits[NC][32] + global max (order-preserving uint atomicMax)
__global__ __launch_bounds__(128)
void k_gvp_scalar(const float* __restrict__ s_in, const float* __restrict__ v_in,
                  const half_t* __restrict__ wh, const half_t* __restrict__ ws_w,
                  const float* __restrict__ ws_b,
                  float* __restrict__ logits, unsigned* __restrict__ gmax)
{
  extern __shared__ unsigned char smem[];
  int wave = threadIdx.x >> 5;
  float*  vh  = (float*)(smem + (size_t)wave * SCALAR_WAVE_BYTES);
  half_t* sXh = (half_t*)(vh + 1536);
  half_t* vth = sXh + 2048;
  half_t* vnh = vth + 1536;
  unsigned* bmax = (unsigned*)(smem + 4*SCALAR_WAVE_BYTES);

  for (int i = threadIdx.x; i < 512; i += 128) bmax[i] = 0x007FFFFFu;
  __syncthreads();

  int row0 = (blockIdx.x * 4 + wave) * 16;    // 16 rows = all 16 channels of one node
  stage_sv(s_in, v_in, nullptr, row0, sXh, vth);
  compute_vh_vn(vth, wh, vh, vnh, nullptr);

  v8f acc[2]; acc[0] = (v8f){}; acc[1] = (v8f){};
#pragma unroll
  for (int kc = 0; kc < 5; ++kc) {
    v16h a = (kc < 4) ? loadA_f16(sXh, 128, 0, kc*32) : loadA_f16(vnh, 32, 0, 0);
#pragma unroll
    for (int nt = 0; nt < 2; ++nt)
      acc[nt] = wmma_f16(a, loadB_Wt(ws_w, 160, nt*16, kc*32), acc[nt]);
  }
#pragma unroll
  for (int nt = 0; nt < 2; ++nt) {
    int n = nt*16 + (lane_id() & 15);
    int mb = (lane_id() & 16) ? 8 : 0;
    float bn = ws_b[n];
#pragma unroll
    for (int i = 0; i < 8; ++i) {
      int r = mb + i;                         // local row == channel c
      float val = acc[nt][i] + bn;
      logits[(size_t)(row0 + r)*32 + n] = val;
      atomicMax(&bmax[r*32 + n], fenc(val));
    }
  }
  __syncthreads();
  for (int i = threadIdx.x; i < 512; i += 128) atomicMax(&gmax[i], bmax[i]);
}

// per-wave region: f32 vh[48][32], so[16][128], vo[48][32], gt[16][32];
// f16 sXh[16][128], vth[48][32], vh16[48][32], vnh[16][32], sgh[16][128]
#define FULL_WAVE_BYTES ((1536+2048+1536+512)*4 + (2048+1536+1536+512+2048)*2)  /* 37888 */

// full GVP (silu/sigmoid, vector gate): f32 in -> f16 out
__global__ __launch_bounds__(128)
void k_gvp_full(const float* __restrict__ s_in, const float* __restrict__ v_in,
                const float* __restrict__ Zbuf,
                const half_t* __restrict__ wh, const half_t* __restrict__ ws_w,
                const float* __restrict__ ws_b,
                const half_t* __restrict__ wv, const half_t* __restrict__ wsv_w,
                const float* __restrict__ wsv_b,
                half_t* __restrict__ s_outh, half_t* __restrict__ v_outh)
{
  extern __shared__ unsigned char smem[];
  int wave = threadIdx.x >> 5;
  float*  vh   = (float*)(smem + (size_t)wave * FULL_WAVE_BYTES);
  float*  so   = vh + 1536;
  float*  vo   = so + 2048;
  float*  gt   = vo + 1536;
  half_t* sXh  = (half_t*)(gt + 512);
  half_t* vth  = sXh + 2048;
  half_t* vh16 = vth + 1536;
  half_t* vnh  = vh16 + 1536;
  half_t* sgh  = vnh + 512;
  int row0 = (blockIdx.x * 4 + wave) * 16;

  stage_sv(s_in, v_in, Zbuf, row0, sXh, vth);
  compute_vh_vn(vth, wh, vh, vnh, vh16);

  // s_out = X @ ws_w^T + b : M16 x K160 x N128 ; A hoisted across 8 N-tiles
  {
    v8f acc[8];
#pragma unroll
    for (int nt = 0; nt < 8; ++nt) acc[nt] = (v8f){};
#pragma unroll
    for (int kc = 0; kc < 5; ++kc) {
      v16h a = (kc < 4) ? loadA_f16(sXh, 128, 0, kc*32) : loadA_f16(vnh, 32, 0, 0);
#pragma unroll
      for (int nt = 0; nt < 8; ++nt)
        acc[nt] = wmma_f16(a, loadB_Wt(ws_w, 160, nt*16, kc*32), acc[nt]);
    }
#pragma unroll
    for (int nt = 0; nt < 8; ++nt) storeD_bias(so, 128, 0, nt*16, acc[nt], ws_b);
  }
  lds_fence();

  // sigmoid(s_out) staged once to f16
  for (int i = lane_id(); i < 16*128; i += 32)
    sgh[i] = (_Float16)(1.0f/(1.0f + __expf(-so[i])));
  lds_fence();

  // gate = sigmoid(s_out) @ wsv^T + b : M16 x K128 x N32
  {
    v8f acc[2]; acc[0] = (v8f){}; acc[1] = (v8f){};
#pragma unroll
    for (int kc = 0; kc < 4; ++kc) {
      v16h a = loadA_f16(sgh, 128, 0, kc*32);
#pragma unroll
      for (int nt = 0; nt < 2; ++nt)
        acc[nt] = wmma_f16(a, loadB_Wt(wsv_w, 128, nt*16, kc*32), acc[nt]);
    }
#pragma unroll
    for (int nt = 0; nt < 2; ++nt) storeD_bias(gt, 32, 0, nt*16, acc[nt], wsv_b);
  }

  // v_out(pre-gate) = vh @ wv^T : M48 x K32 x N32
#pragma unroll
  for (int mt = 0; mt < 3; ++mt) {
    v16h a = loadA_f16(vh16, 32, mt*16, 0);
#pragma unroll
    for (int nt = 0; nt < 2; ++nt) {
      v8f acc = {};
      acc = wmma_f16(a, loadB_Wt(wv, 32, nt*16, 0), acc);
      storeD(vo, 32, mt*16, nt*16, acc);
    }
  }
  lds_fence();

  for (int i = lane_id(); i < 16*128; i += 32) {
    int r = i >> 7;
    float x = so[i];
    s_outh[(size_t)(row0 + r)*128 + (i & 127)] = (_Float16)(x / (1.0f + __expf(-x)));  // silu
  }
  for (int i = lane_id(); i < 16*96; i += 32) {
    int r = i / 96, f = i - r*96;
    int vix = f / 3, d = f - vix*3;
    float g = gt[r*32 + vix];
    float val = vo[(r*3 + d)*32 + vix] * (1.0f/(1.0f + __expf(-g)));
    v_outh[(size_t)(row0 + r)*96 + f] = (_Float16)val;
  }
}

// anchor-weighted pooling partials: for fixed c and 128-node chunk, GEMMs w^T@s, w^T@v, Sum(w)
// part layout per (chunk,c): [0,4096) Sg a*128+s ; [4096,7168) Vg a*96+f ; [7168,7200) Z
// wave w owns row-tile mt=w&1 and column-tiles nt = (w>>1)+4j (14 column tiles: 8 Sg + 6 Vg)
__global__ __launch_bounds__(256)
void k_compress(const float* __restrict__ s_in, const float* __restrict__ v_in,
                const float* __restrict__ logits, const unsigned* __restrict__ gmax,
                float* __restrict__ part)
{
  extern __shared__ unsigned char smem[];
  half_t* wL = (half_t*)smem;        // [128][32]
  half_t* sL = wL + 128*32;          // [128][128]
  half_t* vL = sL + 128*128;         // [128][96]
  int c = blockIdx.y;
  int n0 = blockIdx.x * 128;
  float* myPart = part + (size_t)(blockIdx.x*16 + c) * 7200;

  for (int i = threadIdx.x; i < 128*32; i += 256) {
    int nd = i >> 5, a = i & 31;
    float lg = logits[((size_t)(n0 + nd)*16 + c)*32 + a];
    wL[i] = (_Float16)__expf(lg - fdec(gmax[c*32 + a]));
  }
  for (int i = threadIdx.x; i < 128*128; i += 256) {
    int nd = i >> 7;
    sL[i] = (_Float16)s_in[((size_t)(n0 + nd)*16 + c)*128 + (i & 127)];
  }
  for (int i = threadIdx.x; i < 128*96; i += 256) {
    int nd = i / 96;
    vL[i] = (_Float16)v_in[((size_t)(n0 + nd)*16 + c)*96 + (i - nd*96)];
  }
  __syncthreads();

  int wave = threadIdx.x >> 5, l = lane_id();
  if (wave == 0) {                       // Z partial: lane = anchor
    float zs = 0.0f;
    for (int nd = 0; nd < 128; ++nd) zs += (float)wL[nd*32 + l];
    myPart[7168 + l] = zs;
  }

  int mt  = wave & 1;
  int ntb = wave >> 1;                   // 0..3
  v8f acc[4];
#pragma unroll
  for (int j = 0; j < 4; ++j) acc[j] = (v8f){};
#pragma unroll
  for (int kc = 0; kc < 4; ++kc) {
    v16h a = loadA_f16T(wL, 32, mt*16, kc*32);   // A[a][node] = w[node][a]
#pragma unroll
    for (int j = 0; j < 4; ++j) {
      int nt = ntb + 4*j;
      if (nt < 14) {
        v16h b = (nt < 8) ? loadB_kn(sL, 128, kc*32, nt*16)
                          : loadB_kn(vL, 96,  kc*32, (nt - 8)*16);
        acc[j] = wmma_f16(a, b, acc[j]);
      }
    }
  }
#pragma unroll
  for (int j = 0; j < 4; ++j) {
    int nt = ntb + 4*j;
    if (nt < 14) {
      int mb = mt*16 + ((l & 16) ? 8 : 0);
      if (nt < 8) {
        int n = nt*16 + (l & 15);
#pragma unroll
        for (int i = 0; i < 8; ++i) myPart[(mb + i)*128 + n] = acc[j][i];
      } else {
        int n = (nt - 8)*16 + (l & 15);
#pragma unroll
        for (int i = 0; i < 8; ++i) myPart[4096 + (mb + i)*96 + n] = acc[j][i];
      }
    }
  }
}

__global__ void k_reduce(const float* __restrict__ part, float* __restrict__ Sg,
                         float* __restrict__ Vg, float* __restrict__ Z)
{
  int idx = blockIdx.x*256 + threadIdx.x;
  if (idx >= 16*7200) return;
  int c = idx / 7200, j = idx - c*7200;
  float sum = 0.0f;
  for (int chk = 0; chk < 64; ++chk) sum += part[(size_t)(chk*16 + c)*7200 + j];
  if (j < 4096)      { int a = j >> 7; Sg[((size_t)a*16 + c)*128 + (j & 127)] = sum; }
  else if (j < 7168) { int u = j - 4096; int a = u / 96; Vg[((size_t)a*16 + c)*96 + (u - a*96)] = sum; }
  else               Z[c*32 + (j - 7168)] = sum;
}

// ---------------- attention ----------------
INL v16h loadA_qfeat(const half_t* sqh, const half_t* vqh, int node0, int c, int h, int k0) {
  int l = lane_id();
  int m = l & 15;
  int kb = k0 + ((l & 16) ? 16 : 0);
  size_t row = (size_t)(node0 + m)*16 + c;
  const half_t* sp = sqh + row*128 + h*32;
  const half_t* vp = vqh + row*96  + h*24;
  v16h a;
#pragma unroll
  for (int j = 0; j < 8; ++j) {
    int k1 = kb + j, k2 = kb + 8 + j;
    a[j]     = (k1 < 32) ? sp[k1] : ((k1 < 56) ? vp[k1 - 32] : (_Float16)0.0f);
    a[j + 8] = (k2 < 32) ? sp[k2] : ((k2 < 56) ? vp[k2 - 32] : (_Float16)0.0f);
  }
  return a;
}

INL v16h loadB_kfeat(const half_t* skh, const half_t* vkh, int c, int h, int n0, int k0) {
  int l = lane_id();
  int r = n0 + (l & 15);
  int kb = k0 + ((l & 16) ? 16 : 0);
  const half_t* sp = skh + ((size_t)r*16 + c)*128 + h*32;
  const half_t* vp = vkh + ((size_t)r*16 + c)*96  + h*24;
  v16h b;
#pragma unroll
  for (int j = 0; j < 16; ++j) {
    int k = kb + j;
    b[j] = (k < 32) ? sp[k] : ((k < 56) ? vp[k - 32] : (_Float16)0.0f);
  }
  return b;
}

INL v16h loadB_vfeat(const half_t* svh, const half_t* vvh, int c, int h, int n0) {
  int l = lane_id();
  int f = n0 + (l & 15);
  int kb = (l & 16) ? 16 : 0;
  v16h b;
#pragma unroll
  for (int j = 0; j < 16; ++j) {
    int r = kb + j;
    if (f < 32)      b[j] = svh[((size_t)r*16 + c)*128 + h*32 + f];
    else if (f < 56) b[j] = vvh[((size_t)r*16 + c)*96  + h*24 + (f - 32)];
    else             b[j] = (_Float16)0.0f;
  }
  return b;
}

__global__ __launch_bounds__(128)
void k_attn(const half_t* __restrict__ sqh, const half_t* __restrict__ vqh,
            const half_t* __restrict__ skh, const half_t* __restrict__ vkh,
            const half_t* __restrict__ svh, const half_t* __restrict__ vvh,
            float* __restrict__ out_s, float* __restrict__ out_v)
{
  extern __shared__ unsigned char smem[];
  int wave = threadIdx.x >> 5, l = lane_id();
  float* e = (float*)smem + wave*512;          // [16 nodes][32 anchors]
  int c = blockIdx.y;
  int node0 = (blockIdx.x * 4 + wave) * 16;
  const float escale = 0.15811388f;            // 1/sqrt(40)

  for (int h = 0; h < 4; ++h) {
    // energy = Q @ K^T : M16 x K56(pad64) x N32 ; A hoisted across both r-tiles
    {
      v8f acc[2]; acc[0] = (v8f){}; acc[1] = (v8f){};
#pragma unroll
      for (int kc = 0; kc < 2; ++kc) {
        v16h a = loadA_qfeat(sqh, vqh, node0, c, h, kc*32);
#pragma unroll
        for (int rt = 0; rt < 2; ++rt)
          acc[rt] = wmma_f16(a, loadB_kfeat(skh, vkh, c, h, rt*16, kc*32), acc[rt]);
      }
#pragma unroll
      for (int rt = 0; rt < 2; ++rt) {
        int n = rt*16 + (l & 15);
        int mb = (l & 16) ? 8 : 0;
#pragma unroll
        for (int i = 0; i < 8; ++i) e[(mb + i)*32 + n] = acc[rt][i] * escale;
      }
    }
    lds_fence();
    if (l < 16) {                               // softmax over 32 anchors, row = node
      float mx = -1e30f;
      for (int r2 = 0; r2 < 32; ++r2) mx = fmaxf(mx, e[l*32 + r2]);
      float sum = 0.0f;
      for (int r2 = 0; r2 < 32; ++r2) { float w = __expf(e[l*32 + r2] - mx); e[l*32 + r2] = w; sum += w; }
      float inv = 1.0f / sum;
      for (int r2 = 0; r2 < 32; ++r2) e[l*32 + r2] *= inv;
    }
    lds_fence();
    // out = alpha @ Vfeat : M16 x K32 x N56(pad64) ; alpha A built once
    {
      v16h aA = loadA_f32(e, 32, 0, 0);
#pragma unroll
      for (int ft = 0; ft < 4; ++ft) {
        v8f acc = {};
        acc = wmma_f16(aA, loadB_vfeat(svh, vvh, c, h, ft*16), acc);
        int f = ft*16 + (l & 15);
        int mb = (l & 16) ? 8 : 0;
#pragma unroll
        for (int i = 0; i < 8; ++i) {
          size_t row = (size_t)(node0 + mb + i)*16 + c;
          if (f < 32)      out_s[row*128 + h*32 + f] = acc[i];
          else if (f < 56) out_v[row*96  + h*24 + (f - 32)] = acc[i];
        }
      }
    }
    lds_fence();
  }
}

// ---------------- host ----------------
extern "C" void kernel_launch(void* const* d_in, const int* in_sizes, int n_in,
                              void* d_out, int out_size, void* d_ws, size_t ws_size,
                              hipStream_t stream)
{
  (void)in_sizes; (void)n_in; (void)out_size; (void)ws_size;
  const float* s        = (const float*)d_in[0];
  const float* v        = (const float*)d_in[1];
  const float* wp_wh    = (const float*)d_in[2];
  const float* wp_ws_w  = (const float*)d_in[3];
  const float* wp_ws_b  = (const float*)d_in[4];
  const float* q_wh     = (const float*)d_in[5];
  const float* q_ws_w   = (const float*)d_in[6];
  const float* q_ws_b   = (const float*)d_in[7];
  const float* q_wv     = (const float*)d_in[8];
  const float* q_wsv_w  = (const float*)d_in[9];
  const float* q_wsv_b  = (const float*)d_in[10];
  const float* k_wh     = (const float*)d_in[11];
  const float* k_ws_w   = (const float*)d_in[12];
  const float* k_ws_b   = (const float*)d_in[13];
  const float* k_wv     = (const float*)d_in[14];
  const float* k_wsv_w  = (const float*)d_in[15];
  const float* k_wsv_b  = (const float*)d_in[16];
  const float* p_wh     = (const float*)d_in[17];
  const float* p_ws_w   = (const float*)d_in[18];
  const float* p_ws_b   = (const float*)d_in[19];
  const float* p_wv     = (const float*)d_in[20];
  const float* p_wsv_w  = (const float*)d_in[21];
  const float* p_wsv_b  = (const float*)d_in[22];

  unsigned char* ws = (unsigned char*)d_ws;
  size_t off = 0;
  auto alloc = [&](size_t bytes) -> void* {
    off = (off + 255) & ~(size_t)255;
    void* p = ws + off;
    off += bytes;
    return p;
  };

  half_t* wp_wh_h  = (half_t*)alloc(1024*2);
  half_t* wp_ws_h  = (half_t*)alloc(5120*2);
  half_t* q_wh_h   = (half_t*)alloc(1024*2);
  half_t* q_ws_h   = (half_t*)alloc(20480*2);
  half_t* q_wv_h   = (half_t*)alloc(1024*2);
  half_t* q_wsv_h  = (half_t*)alloc(4096*2);
  half_t* k_wh_h   = (half_t*)alloc(1024*2);
  half_t* k_ws_h   = (half_t*)alloc(20480*2);
  half_t* k_wv_h   = (half_t*)alloc(1024*2);
  half_t* k_wsv_h  = (half_t*)alloc(4096*2);
  half_t* p_wh_h   = (half_t*)alloc(1024*2);
  half_t* p_ws_h   = (half_t*)alloc(20480*2);
  half_t* p_wv_h   = (half_t*)alloc(1024*2);
  half_t* p_wsv_h  = (half_t*)alloc(4096*2);
  float*    logits = (float*)alloc((size_t)NCROWS*32*4);
  unsigned* gmax   = (unsigned*)alloc(512*4);
  float*    Z      = (float*)alloc(512*4);
  float*    Sg     = (float*)alloc(65536*4);
  float*    Vg     = (float*)alloc(49152*4);
  float*    part   = (float*)alloc((size_t)64*16*7200*4);
  half_t*   sqh    = (half_t*)alloc((size_t)NCROWS*128*2);
  half_t*   vqh    = (half_t*)alloc((size_t)NCROWS*96*2);
  half_t*   skh    = (half_t*)alloc(512*128*2);
  half_t*   vkh    = (half_t*)alloc(512*96*2);
  half_t*   svh    = (half_t*)alloc(512*128*2);
  half_t*   vvh    = (half_t*)alloc(512*96*2);

  float* out_s = (float*)d_out;
  float* out_v = out_s + (size_t)NCROWS*128;

  auto cvt = [&](const float* srcp, half_t* dstp, int n) {
    k_cvt<<<(n + 255)/256, 256, 0, stream>>>(srcp, dstp, n);
  };
  cvt(wp_wh, wp_wh_h, 1024);  cvt(wp_ws_w, wp_ws_h, 5120);
  cvt(q_wh,  q_wh_h,  1024);  cvt(q_ws_w,  q_ws_h, 20480);
  cvt(q_wv,  q_wv_h,  1024);  cvt(q_wsv_w, q_wsv_h, 4096);
  cvt(k_wh,  k_wh_h,  1024);  cvt(k_ws_w,  k_ws_h, 20480);
  cvt(k_wv,  k_wv_h,  1024);  cvt(k_wsv_w, k_wsv_h, 4096);
  cvt(p_wh,  p_wh_h,  1024);  cvt(p_ws_w,  p_ws_h, 20480);
  cvt(p_wv,  p_wv_h,  1024);  cvt(p_wsv_w, p_wsv_h, 4096);
  k_initmax<<<2, 256, 0, stream>>>(gmax);

  const size_t smemScalar = 4*SCALAR_WAVE_BYTES + 512*4;   // 59392
  const size_t smemFull   = 4*FULL_WAVE_BYTES;             // 151552
  const size_t smemComp   = (4096 + 16384 + 12288)*2;      // 65536
  const size_t smemAttn   = 4*512*4;                       // 8192

  k_gvp_scalar<<<NCROWS/64, 128, smemScalar, stream>>>(s, v, wp_wh_h, wp_ws_h, wp_ws_b,
                                                       logits, gmax);
  k_gvp_full<<<NCROWS/64, 128, smemFull, stream>>>(s, v, nullptr,
                                                   q_wh_h, q_ws_h, q_ws_b, q_wv_h, q_wsv_h, q_wsv_b,
                                                   sqh, vqh);
  k_compress<<<dim3(64, 16), 256, smemComp, stream>>>(s, v, logits, gmax, part);
  k_reduce<<<(16*7200 + 255)/256, 256, 0, stream>>>(part, Sg, Vg, Z);
  k_gvp_full<<<512/64, 128, smemFull, stream>>>(Sg, Vg, Z,
                                                k_wh_h, k_ws_h, k_ws_b, k_wv_h, k_wsv_h, k_wsv_b,
                                                skh, vkh);
  k_gvp_full<<<512/64, 128, smemFull, stream>>>(Sg, Vg, Z,
                                                p_wh_h, p_ws_h, p_ws_b, p_wv_h, p_wsv_h, p_wsv_b,
                                                svh, vvh);
  k_attn<<<dim3(N_NODES/64, 16), 128, smemAttn, stream>>>(sqh, vqh, skh, vkh, svh, vvh,
                                                          out_s, out_v);
}
